// RNNFrame_57647051046974
// MI455X (gfx1250) — matrix operand order repaired and checked by reading, not compiled, not run on hardware
//
#include <hip/hip_runtime.h>
#include <hip/hip_bf16.h>
#include <stdint.h>

// ---------------------------------------------------------------------------
// Problem constants (match reference)
// ---------------------------------------------------------------------------
#define TT 512
#define BB 64
#define DD 512
#define HH 512
#define G4 2048   // 4*H

typedef __attribute__((ext_vector_type(16))) __bf16 v16bf;
typedef __attribute__((ext_vector_type(8)))  float  v8f;

union FragBF {
    v16bf v;
    uint4 q[2];
};
union Pack8BF {
    uint4 q;
    uint16_t u[8];
};

__device__ __forceinline__ v8f vzero8() {
    v8f z;
#pragma unroll
    for (int i = 0; i < 8; ++i) z[i] = 0.0f;
    return z;
}

__device__ __forceinline__ uint16_t f2bf(float f) {
    uint32_t u = __float_as_uint(f);
    uint32_t r = (u + 0x7FFFu + ((u >> 16) & 1u)) >> 16;  // round-nearest-even
    return (uint16_t)r;
}
__device__ __forceinline__ float bf2f(uint16_t h) {
    return __uint_as_float(((uint32_t)h) << 16);
}
// sigmoid(x) = rcp(1 + exp(-x))         : v_exp_f32 + v_rcp_f32
__device__ __forceinline__ float fast_sigmoid(float x) {
    return __builtin_amdgcn_rcpf(1.0f + __expf(-x));
}
// tanh(x) = (1 - e) * rcp(1 + e), e = exp(-2x)
__device__ __forceinline__ float fast_tanh(float x) {
    float e = __expf(-2.0f * x);
    return (1.0f - e) * __builtin_amdgcn_rcpf(1.0f + e);
}

// A fragment: 16x32 bf16, row-major source (rows m0..m0+15, cols k0..k0+31).
// Lane layout (ISA 7.12.2): lanes 0-15 hold K {k0..k0+7, k0+16..k0+23},
// lanes 16-31 hold K {k0+8..k0+15, k0+24..k0+31}; two contiguous 16B chunks.
__device__ __forceinline__ v16bf load_a16(const uint16_t* base, int ld,
                                          int m0, int k0, int lane) {
    int half = lane >> 4;
    int r    = lane & 15;
    const uint16_t* p = base + (size_t)(m0 + r) * ld + k0 + half * 8;
    FragBF f;
    f.q[0] = *(const uint4*)(p);
    f.q[1] = *(const uint4*)(p + 16);
    return f.v;
}

// B fragment: 32x16 bf16 (KxN). Source is W row-major [N, K] (i.e. B = W^T).
// Lane = column n (mod 16); lanes 0-15 hold K {k0..k0+7, k0+16..k0+23}, lanes 16-31 K+8.
// Two 16B chunks per lane from row n of W.
__device__ __forceinline__ v16bf load_b16(const uint16_t* base, int ld,
                                          int n0, int k0, int lane) {
    int half = lane >> 4;
    int c    = lane & 15;
    const uint16_t* p = base + (size_t)(n0 + c) * ld + k0 + half * 8;
    FragBF f;
    f.q[0] = *(const uint4*)(p);
    f.q[1] = *(const uint4*)(p + 16);
    return f.v;
}

__device__ __forceinline__ v8f wmma_bf16(v16bf a, v16bf b, v8f c) {
    return __builtin_amdgcn_wmma_f32_16x16x32_bf16(
        /*neg_a=*/false, a, /*neg_b=*/false, b,
        /*c_mod=*/(short)0, c, /*reuse_a=*/false, /*reuse_b=*/false);
}

// ---------------------------------------------------------------------------
// fp32 -> bf16 conversion (grid-stride)
// ---------------------------------------------------------------------------
__global__ void k_f32_to_bf16(const float* __restrict__ src,
                              uint16_t* __restrict__ dst, size_t n) {
    size_t i = (size_t)blockIdx.x * blockDim.x + threadIdx.x;
    size_t stride = (size_t)gridDim.x * blockDim.x;
    for (; i < n; i += stride) dst[i] = f2bf(src[i]);
}

// ---------------------------------------------------------------------------
// gx[dir] = X (bf16 [M,K]) @ W[dir]^T (bf16 [N,K] row-major) + bias[dir][N]
// Output bf16, TIME-MAJOR TRANSPOSED: gx[t][n (4H)][b (B)] so the scan can
// read its 8 contiguous batch rows with one b128 load per gate tile.
// Grid: (N/64, M/128, 2 dirs). Block: 256 (8 waves), wave = 32x32 output.
// Each wave's 32 M-rows lie inside one 64-row (= one timestep) block.
// ---------------------------------------------------------------------------
__global__ void k_gemm_bias(const uint16_t* __restrict__ X,
                            const uint16_t* __restrict__ Wall,
                            const float* __restrict__ biasAll,
                            uint16_t* __restrict__ Yall,
                            int M, int N, int K) {
    const int dir = blockIdx.z;
    const uint16_t* W = Wall + (size_t)dir * N * K;
    const float* bias = biasAll + (size_t)dir * N;
    uint16_t* Y = Yall + (size_t)dir * M * N;

    const int lane = threadIdx.x & 31;
    const int wave = threadIdx.x >> 5;          // 0..7
    const int wm = wave & 3;                    // 4 M sub-blocks
    const int wn = wave >> 2;                   // 2 N sub-blocks
    const int mbase = blockIdx.y * 128 + wm * 32;
    const int nbase = blockIdx.x * 64 + wn * 32;

    v8f acc00 = vzero8(), acc01 = vzero8(), acc10 = vzero8(), acc11 = vzero8();

    for (int k0 = 0; k0 < K; k0 += 32) {
        v16bf a0 = load_a16(X, K, mbase,      k0, lane);
        v16bf a1 = load_a16(X, K, mbase + 16, k0, lane);
        v16bf b0 = load_b16(W, K, nbase,      k0, lane);
        v16bf b1 = load_b16(W, K, nbase + 16, k0, lane);
        acc00 = wmma_bf16(a0, b0, acc00);
        acc01 = wmma_bf16(a0, b1, acc01);
        acc10 = wmma_bf16(a1, b0, acc10);
        acc11 = wmma_bf16(a1, b1, acc11);
    }

    // Transposed store: one b128 (8 packed bf16 over batch) per tile.
    const int rowoff = (lane >> 4) * 8;
    const int c = lane & 15;
    const int t = mbase >> 6;                   // timestep of this wave
    const int bcol0 = (mbase & 63) + rowoff;    // batch column base (mult of 8)
    uint16_t* Yt = Y + (size_t)t * G4 * BB;
#pragma unroll
    for (int nt = 0; nt < 2; ++nt) {
        int n = nbase + nt * 16 + c;
        float bn = bias[n];
#pragma unroll
        for (int mt = 0; mt < 2; ++mt) {
            v8f acc = (mt == 0) ? (nt == 0 ? acc00 : acc01)
                                : (nt == 0 ? acc10 : acc11);
            Pack8BF pk;
#pragma unroll
            for (int r = 0; r < 8; ++r) pk.u[r] = f2bf(acc[r] + bn);
            *(uint4*)(Yt + (size_t)n * BB + bcol0 + mt * 16) = pk.q;
        }
    }
}

// ---------------------------------------------------------------------------
// LSTM sequential scan, one direction per block. Block = 1024 threads
// (32 waves). h double-buffered bf16 in LDS (128 KB); c in registers.
// Wave w owns 16 columns j0 = w*16 of H across all 4 gates and 4 M tiles.
// gx: bf16 [2][T][4H][B] (transposed; time-indexed by ORIGINAL t).
// Wh: bf16 [2][4H*H] row-major.
// layer0: out_bf = x1 bf16 [T*B, 2H]; layer1: out_f32 = o1 fp32 [T*B, 2H].
// ---------------------------------------------------------------------------
__global__ void k_lstm_scan(const uint16_t* __restrict__ gxAll,
                            const uint16_t* __restrict__ WhAll,
                            uint16_t* __restrict__ out_bf,
                            float* __restrict__ out_f32,
                            float* __restrict__ hn,
                            float* __restrict__ cn,
                            int useF32Out) {
    __shared__ uint16_t hbuf[2][BB * HH];  // 2 * 64 * 512 * 2B = 128 KB

    const int dir = blockIdx.x;
    const uint16_t* gx = gxAll + (size_t)dir * TT * BB * G4;
    const uint16_t* Wh = WhAll + (size_t)dir * G4 * HH;

    const int lane = threadIdx.x & 31;
    const int wave = threadIdx.x >> 5;   // 0..31
    const int j0 = wave * 16;            // column block in [0,512)
    const int c = lane & 15;
    const int rowoff = (lane >> 4) * 8;
    const int n = j0 + c;                // this lane's H-column

    // zero h[0]
    {
        uint32_t* zp = (uint32_t*)&hbuf[0][0];
        for (int i = threadIdx.x; i < BB * HH / 2; i += blockDim.x) zp[i] = 0u;
    }
    float creg[4][8];
#pragma unroll
    for (int mt = 0; mt < 4; ++mt)
#pragma unroll
        for (int r = 0; r < 8; ++r) creg[mt][r] = 0.0f;

    __syncthreads();

    for (int s = 0; s < TT; ++s) {
        const int t = (dir == 0) ? s : (TT - 1 - s);
        const uint16_t* hprev = hbuf[s & 1];
        uint16_t* hnext = hbuf[(s + 1) & 1];
        const uint16_t* gxt = gx + (size_t)t * G4 * BB;  // [4H][B]

#pragma unroll 1
        for (int mt = 0; mt < 4; ++mt) {
            const int m0 = mt * 16;
            const int mb = m0 + rowoff;         // this lane's 8-row batch base
            v8f acc[4];
#pragma unroll
            for (int g = 0; g < 4; ++g) acc[g] = vzero8();

            for (int k0 = 0; k0 < HH; k0 += 32) {
                v16bf a = load_a16(hprev, HH, m0, k0, lane);  // ds loads
#pragma unroll
                for (int g = 0; g < 4; ++g) {
                    v16bf b = load_b16(Wh, HH, g * HH + j0, k0, lane);
                    acc[g] = wmma_bf16(a, b, acc[g]);
                }
            }

            // gx for this lane: 8 contiguous batch entries per gate (b128 each)
            Pack8BF gk[4];
#pragma unroll
            for (int g = 0; g < 4; ++g)
                gk[g].q = *(const uint4*)(gxt + (size_t)(g * HH + n) * BB + mb);

            // fused i/f/g/o elementwise + state update
#pragma unroll
            for (int r = 0; r < 8; ++r) {
                const int m = mb + r;
                float gi = acc[0][r] + bf2f(gk[0].u[r]);
                float gf = acc[1][r] + bf2f(gk[1].u[r]);
                float gg = acc[2][r] + bf2f(gk[2].u[r]);
                float go = acc[3][r] + bf2f(gk[3].u[r]);
                float iv = fast_sigmoid(gi);
                float fv = fast_sigmoid(gf);
                float ov = fast_sigmoid(go);
                float gv = fast_tanh(gg);
                float cv = fv * creg[mt][r] + iv * gv;
                creg[mt][r] = cv;
                float hv = ov * fast_tanh(cv);
                uint16_t hb = f2bf(hv);

                hnext[m * HH + n] = hb;  // ds store

                size_t oidx = ((size_t)t * BB + m) * (2 * HH) + (size_t)dir * HH + n;
                if (useF32Out) out_f32[oidx] = hv;
                else           out_bf[oidx] = hb;

                if (s == TT - 1) {
                    size_t sidx = (size_t)dir * BB * HH + (size_t)m * HH + n;
                    hn[sidx] = hv;
                    cn[sidx] = cv;
                }
            }
        }
        __syncthreads();  // publish h(t) before anyone reads it at step s+1
    }
}

// ---------------------------------------------------------------------------
// Host launcher
// ---------------------------------------------------------------------------
extern "C" void kernel_launch(void* const* d_in, const int* in_sizes, int n_in,
                              void* d_out, int out_size, void* d_ws, size_t ws_size,
                              hipStream_t stream) {
    (void)in_sizes; (void)n_in; (void)out_size; (void)ws_size;

    const float* x     = (const float*)d_in[0];  // [T,B,D]
    const float* Wi_l0 = (const float*)d_in[1];  // [2,4H,D]
    const float* Wh_l0 = (const float*)d_in[2];  // [2,4H,H]
    const float* b_l0  = (const float*)d_in[3];  // [2,4H]
    const float* Wi_l1 = (const float*)d_in[4];  // [2,4H,2H]
    const float* Wh_l1 = (const float*)d_in[5];  // [2,4H,H]
    const float* b_l1  = (const float*)d_in[6];  // [2,4H]
    float* out = (float*)d_out;

    const size_t M = (size_t)TT * BB;            // 32768
    const size_t nXb   = M * DD;                 // x bf16
    const size_t nX1b  = M * 2 * HH;             // layer1 input bf16
    const size_t nWi0  = (size_t)2 * G4 * DD;
    const size_t nWh0  = (size_t)2 * G4 * HH;
    const size_t nWi1  = (size_t)2 * G4 * (2 * HH);
    const size_t nWh1  = (size_t)2 * G4 * HH;
    const size_t nGx   = (size_t)2 * M * G4;     // both dirs, reused per layer

    char* ws = (char*)d_ws;
    size_t off = 0;
    auto alloc_u16 = [&](size_t nelem) {
        uint16_t* p = (uint16_t*)(ws + off);
        off += ((nelem * 2 + 255) & ~(size_t)255);
        return p;
    };
    uint16_t* xb   = alloc_u16(nXb);
    uint16_t* x1b  = alloc_u16(nX1b);
    uint16_t* Wib0 = alloc_u16(nWi0);
    uint16_t* Whb0 = alloc_u16(nWh0);
    uint16_t* Wib1 = alloc_u16(nWi1);
    uint16_t* Whb1 = alloc_u16(nWh1);
    uint16_t* gx   = alloc_u16(nGx);

    // --- conversions to bf16 ---
    auto conv = [&](const float* s, uint16_t* d, size_t nelem) {
        int blocks = (int)((nelem + 1023) / 1024);
        if (blocks > 8192) blocks = 8192;
        k_f32_to_bf16<<<blocks, 256, 0, stream>>>(s, d, nelem);
    };
    conv(x,     xb,   nXb);
    conv(Wi_l0, Wib0, nWi0);
    conv(Wh_l0, Whb0, nWh0);
    conv(Wi_l1, Wib1, nWi1);
    conv(Wh_l1, Whb1, nWh1);

    // d_out tails
    const size_t o1n = M * 2 * HH;                 // o1 elements
    float* hn_base = out + o1n;                    // [4,B,H]
    float* cn_base = hn_base + (size_t)4 * BB * HH;

    // --- layer 0 ---
    {
        dim3 grid(G4 / 64, (unsigned)(M / 128), 2);
        k_gemm_bias<<<grid, 256, 0, stream>>>(xb, Wib0, b_l0, gx,
                                              (int)M, G4, DD);
        k_lstm_scan<<<2, 1024, 0, stream>>>(gx, Whb0,
                                            x1b, nullptr,
                                            hn_base + 0, cn_base + 0,
                                            /*useF32Out=*/0);
    }

    // --- layer 1 ---
    {
        dim3 grid(G4 / 64, (unsigned)(M / 128), 2);
        k_gemm_bias<<<grid, 256, 0, stream>>>(x1b, Wib1, b_l1, gx,
                                              (int)M, G4, 2 * HH);
        k_lstm_scan<<<2, 1024, 0, stream>>>(gx, Whb1,
                                            nullptr, out,
                                            hn_base + (size_t)2 * BB * HH,
                                            cn_base + (size_t)2 * BB * HH,
                                            /*useF32Out=*/1);
    }
}
